// GATLayer_67559835566266
// MI455X (gfx1250) — matrix operand order repaired and checked by reading, not compiled, not run on hardware
//
#include <hip/hip_runtime.h>
#include <hip/hip_bf16.h>

// ---------------------------------------------------------------------------
// GAT layer for MI455X (gfx1250, wave32, WMMA).
//   k1: whT[n][i] = bf16( h @ W^T + b )   (WMMA bf16, f32 accum, L2-resident 4MB)
//   k2: s_src[i], s_dst[i] = whT . a_src/a_dst
//   k3: fused masked-leakyrelu-softmax(attn) @ wh, flash-attention style
//       (online m/l, 16 rows per wave, 32-col tiles, 16x v_wmma per tile)
// ---------------------------------------------------------------------------

typedef __attribute__((ext_vector_type(16))) __bf16          v16bf;
typedef __attribute__((ext_vector_type(16))) unsigned short  v16u;
typedef __attribute__((ext_vector_type(8)))  float           v8f;
typedef __attribute__((ext_vector_type(4)))  float           v4f;
typedef __attribute__((ext_vector_type(4)))  int             v4i;

#define N_NODES 8192
#define D_IN    512
#define D_OUT   256
#define NEG_INF (-9000000000000000.0f)
#define SLOPE   0.2f

// round-to-nearest-even f32 -> bf16 (as raw u16)
__device__ inline unsigned short f2bf(float f) {
    unsigned u = __builtin_bit_cast(unsigned, f);
    u += 0x7FFFu + ((u >> 16) & 1u);
    return (unsigned short)(u >> 16);
}
__device__ inline float bf2f(unsigned short b) {
    return __builtin_bit_cast(float, (unsigned)b << 16);
}

// ---------------------------------------------------------------------------
// Kernel 1: whT (D_OUT x N_NODES, bf16) = (h @ W^T + b)^T
// grid (N/16, 4), block 128 (4 waves); wave w owns n-tile blockIdx.y*4+w.
// ---------------------------------------------------------------------------
__global__ __launch_bounds__(128) void wh_gemm_bf16(
        const float* __restrict__ h, const float* __restrict__ W,
        const float* __restrict__ Wb, unsigned short* __restrict__ whT) {
    const int lane  = threadIdx.x & 31;
    const int wave  = threadIdx.x >> 5;
    const int row   = lane & 15;
    const int hi    = lane >> 4;                 // which half of the wave
    const int ibase = blockIdx.x * 16;           // node tile
    const int nbase = (blockIdx.y * 4 + wave) * 16;
    const int n     = nbase + row;               // B-operand column / output col
    const float bias = Wb[n];

    v8f acc = {};
    for (int kt = 0; kt < D_IN; kt += 32) {
        // A tile: 16x32 of h. lane: M=row; K = kt + hi*16 + 0..15
        const float* ap = h + (size_t)(ibase + row) * D_IN + kt + hi * 16;
        v4f a0 = *(const v4f*)(ap);
        v4f a1 = *(const v4f*)(ap + 4);
        v4f a2 = *(const v4f*)(ap + 8);
        v4f a3 = *(const v4f*)(ap + 12);
        v16u au;
        #pragma unroll
        for (int e = 0; e < 4; ++e) {
            au[e]      = f2bf(a0[e]);
            au[4 + e]  = f2bf(a1[e]);
            au[8 + e]  = f2bf(a2[e]);
            au[12 + e] = f2bf(a3[e]);
        }
        // B tile: 32x16 of W^T. lane: N=row; K = kt + hi*16 + 0..15
        // B[k][n] = W[n][k] -> 16 contiguous floats along k of row n.
        const float* bp = W + (size_t)n * D_IN + kt + hi * 16;
        v16u bu;
        #pragma unroll
        for (int e = 0; e < 16; e += 4) {
            v4f bv4 = *(const v4f*)(bp + e);
            bu[e]     = f2bf(bv4[0]);
            bu[e + 1] = f2bf(bv4[1]);
            bu[e + 2] = f2bf(bv4[2]);
            bu[e + 3] = f2bf(bv4[3]);
        }
        v16bf av = __builtin_bit_cast(v16bf, au);
        v16bf bv = __builtin_bit_cast(v16bf, bu);
        acc = __builtin_amdgcn_wmma_f32_16x16x32_bf16(
                false, av, false, bv, (short)0, acc, false, false);
    }
    // D layout: VGPR r, lane L -> M = r + 8*hi, N = row. Store transposed bf16.
    #pragma unroll
    for (int r = 0; r < 8; ++r) {
        int i = ibase + r + hi * 8;
        whT[(size_t)n * N_NODES + i] = f2bf(acc[r] + bias);
    }
}

// ---------------------------------------------------------------------------
// Kernel 2: s_src / s_dst from whT (coalesced along node index).
// ---------------------------------------------------------------------------
__global__ __launch_bounds__(256) void attn_scores(
        const unsigned short* __restrict__ whT, const float* __restrict__ a_w,
        float* __restrict__ s_src, float* __restrict__ s_dst) {
    const int i = blockIdx.x * 256 + threadIdx.x;
    float ss = 0.f, sd = 0.f;
    #pragma unroll 4
    for (int nf = 0; nf < D_OUT; ++nf) {
        float w = bf2f(whT[(size_t)nf * N_NODES + i]);
        ss += w * a_w[nf];
        sd += w * a_w[D_OUT + nf];
    }
    s_src[i] = ss;
    s_dst[i] = sd;
}

// ---------------------------------------------------------------------------
// Kernel 3: fused masked softmax + attn@wh.
// block 128 (4 waves), each wave owns 16 rows; 32-col tiles, online softmax.
// s_dst staged in LDS (32 KB). A operand = P (16x32 bf16), B = whT tiles.
// ---------------------------------------------------------------------------
__global__ __launch_bounds__(128) void gat_fused_attn(
        const int* __restrict__ adj, const unsigned short* __restrict__ whT,
        const float* __restrict__ s_src, const float* __restrict__ s_dst,
        const float* __restrict__ a_b, float* __restrict__ out) {
    __shared__ float sdst[N_NODES];              // 32 KB of the 320 KB WGP LDS
    for (int t = threadIdx.x; t < N_NODES; t += 128) sdst[t] = s_dst[t];
    __syncthreads();

    const int lane  = threadIdx.x & 31;
    const int wave  = threadIdx.x >> 5;
    const int row   = lane & 15;
    const int hi    = lane >> 4;
    const int ibase = blockIdx.x * 64 + wave * 16;
    const int i     = ibase + row;
    const float ssrc = s_src[i];
    const float ab   = a_b[0];

    float m = NEG_INF, l = 0.f;
    v8f acc[16];
    #pragma unroll
    for (int t = 0; t < 16; ++t) { v8f z = {}; acc[t] = z; }

    const int* adjrow = adj + (size_t)i * N_NODES;

    for (int jt = 0; jt < N_NODES; jt += 32) {
        const int kb = jt + hi * 16;             // this lane's K origin
        v4i m0 = *(const v4i*)(adjrow + kb);
        v4i m1 = *(const v4i*)(adjrow + kb + 4);
        v4i m2 = *(const v4i*)(adjrow + kb + 8);
        v4i m3 = *(const v4i*)(adjrow + kb + 12);
        __builtin_prefetch(adjrow + kb + 32, 0, 0);   // global_prefetch_b8

        // logits for this lane's 16 A-layout elements
        float ev[16];
        #pragma unroll
        for (int e = 0; e < 8; ++e) {
            float x = ssrc + sdst[kb + e] + ab;
            x = x > 0.f ? x : SLOPE * x;
            int a = (e < 4) ? m0[e] : m1[e - 4];
            ev[e] = (a > 0) ? x : NEG_INF;
        }
        #pragma unroll
        for (int e = 0; e < 8; ++e) {
            float x = ssrc + sdst[kb + 8 + e] + ab;
            x = x > 0.f ? x : SLOPE * x;
            int a = (e < 4) ? m2[e] : m3[e - 4];
            ev[8 + e] = (a > 0) ? x : NEG_INF;
        }

        // online softmax: row split across lanes L and L+16
        float mt = ev[0];
        #pragma unroll
        for (int e = 1; e < 16; ++e) mt = fmaxf(mt, ev[e]);
        mt = fmaxf(mt, __shfl_xor(mt, 16, 32));
        float mnew  = fmaxf(m, mt);
        float scale = __expf(m - mnew);

        float ps = 0.f;
        v16u pu;
        #pragma unroll
        for (int e = 0; e < 16; ++e) {
            float p = __expf(ev[e] - mnew);
            ps += p;
            pu[e] = f2bf(p);
        }
        ps += __shfl_xor(ps, 16, 32);
        l = l * scale + ps;
        m = mnew;

        if (scale != 1.0f) {
            #pragma unroll
            for (int t = 0; t < 16; ++t)
                #pragma unroll
                for (int r = 0; r < 8; ++r) acc[t][r] *= scale;
        }

        // acc[t] += P(16x32) @ whT-tile(32x16) for each 16-col output tile t
        v16bf pv = __builtin_bit_cast(v16bf, pu);
        const unsigned short* bbase = whT + (size_t)row * N_NODES + jt + hi * 16;
        #pragma unroll
        for (int t = 0; t < 16; ++t) {
            v16u bu = *(const v16u*)(bbase + (size_t)t * 16 * N_NODES);
            v16bf bv = __builtin_bit_cast(v16bf, bu);
            acc[t] = __builtin_amdgcn_wmma_f32_16x16x32_bf16(
                        false, pv, false, bv, (short)0, acc[t], false, false);
        }
    }

    // epilogue: out[i][:] = acc / l. Lane p (and p+16) holds l for row p.
    #pragma unroll
    for (int r = 0; r < 8; ++r) {
        const int orow = r + hi * 8;
        const float inv = 1.0f / __shfl(l, orow, 32);
        const int gi = ibase + orow;
        #pragma unroll
        for (int t = 0; t < 16; ++t)
            out[(size_t)gi * D_OUT + t * 16 + row] = acc[t][r] * inv;
    }
}

// ---------------------------------------------------------------------------
extern "C" void kernel_launch(void* const* d_in, const int* in_sizes, int n_in,
                              void* d_out, int out_size, void* d_ws, size_t ws_size,
                              hipStream_t stream) {
    const float* h   = (const float*)d_in[0];
    const int*   adj = (const int*)  d_in[1];
    const float* W_w = (const float*)d_in[2];
    const float* W_b = (const float*)d_in[3];
    const float* a_w = (const float*)d_in[4];
    const float* a_b = (const float*)d_in[5];
    float* out = (float*)d_out;

    char* ws = (char*)d_ws;
    unsigned short* whT = (unsigned short*)ws;                       // 4 MB
    float* s_src = (float*)(ws + (size_t)N_NODES * D_OUT * 2);       // 32 KB
    float* s_dst = s_src + N_NODES;                                  // 32 KB

    wh_gemm_bf16 <<<dim3(N_NODES / 16, 4), 128, 0, stream>>>(h, W_w, W_b, whT);
    attn_scores  <<<N_NODES / 256, 256, 0, stream>>>(whT, a_w, s_src, s_dst);
    gat_fused_attn<<<N_NODES / 64, 128, 0, stream>>>(adj, whT, s_src, s_dst, a_b, out);
}
